// VectorizedIPGLayer_67164698575282
// MI455X (gfx1250) — compile-verified, matrix operand order checked by reading, not run on hardware
//
#include <hip/hip_runtime.h>
#include <hip/hip_bf16.h>

#define BB 4
#define CC 64
#define HH 192
#define WW 192
#define HWW (HH*WW)
#define NPIX (BB*HWW)
#define GG 32
#define HD 96

typedef __attribute__((ext_vector_type(16))) __bf16 v16bf;
typedef __attribute__((ext_vector_type(8)))  float  v8f;

union Frag { uint4 q[2]; v16bf v; };

__device__ __forceinline__ unsigned short f2bf(float f) {
  unsigned int u = __float_as_uint(f);
  u += 0x7fffu + ((u >> 16) & 1u);       // round-to-nearest-even
  return (unsigned short)(u >> 16);
}

__device__ __forceinline__ v8f wmma_bf16(v16bf a, v16bf b, v8f c) {
  return __builtin_amdgcn_wmma_f32_16x16x32_bf16(false, a, false, b, (short)0, c, false, false);
}

// ---------------- k0: init min/max slots ----------------
__global__ void k0_init(unsigned int* mm) {
  int t = threadIdx.x;
  if (t < BB) mm[t] = 0x7f800000u;   // +inf  (min slots)
  else if (t < 2*BB) mm[t] = 0u;     // 0     (max slots; df >= 0)
}

// ---------------- k0b: pack weights into per-lane WMMA B fragments ----------------
// B[k][n] layout mirrors A 16-bit 16x32 table: lane holds column n = lane&15,
// vgpr j holds k pair at k0 = (j<4?0:16) + 8*(lane>>4) + 2*(j&3).
__global__ void k0_pack(const float* __restrict__ w1, const float* __restrict__ w2,
                        unsigned int* __restrict__ w1f, unsigned int* __restrict__ w2f) {
  int tid = blockIdx.x * 256 + threadIdx.x;     // 8192 threads
  int sel  = tid >> 12;
  int r    = tid & 4095;
  int t    = r >> 8;          // tile 0..15
  int lane = (r >> 3) & 31;
  int j    = r & 7;
  int half = lane >> 4;
  int l15  = lane & 15;
  int k0 = ((j & 4) ? 16 : 0) + 8*half + 2*(j & 3);
  if (sel == 0) {             // GEMM1: B[k][o] = w1[o][k], w1 is [128,64]
    int kt = t >> 3, ot = t & 7;
    int n = ot*16 + l15;
    int k = kt*32 + k0;
    unsigned int lo = f2bf(w1[n*64 + k]);
    unsigned int hi = f2bf(w1[n*64 + k + 1]);
    w1f[(t*32 + lane)*8 + j] = lo | (hi << 16);
  } else {                    // GEMM2: B[k][c] = w2[c][k], w2 is [64,128]
    int kt = t >> 2, ct = t & 3;
    int n = ct*16 + l15;
    int k = kt*32 + k0;
    unsigned int lo = f2bf(w2[n*128 + k]);
    unsigned int hi = f2bf(w2[n*128 + k + 1]);
    w2f[(t*32 + lane)*8 + j] = lo | (hi << 16);
  }
}

// ---------------- k1: 0.5x bilinear == 2x2 average pool ----------------
__global__ void k1_down(const float* __restrict__ x, float* __restrict__ xd) {
  int tid = blockIdx.x * 256 + threadIdx.x;
  if (tid >= BB*CC*HD*HD) return;
  int xh = tid % HD;
  int yh = (tid / HD) % HD;
  int bc = tid / (HD*HD);
  const float* p = x + (size_t)bc*HWW + (2*yh)*WW + 2*xh;
  xd[tid] = 0.25f * (p[0] + p[1] + p[WW] + p[WW+1]);
}

// ---------------- k2: detail map + per-batch min/max ----------------
__global__ void k2_detail(const float* __restrict__ x, const float* __restrict__ xd,
                          float* __restrict__ df, unsigned int* __restrict__ mm) {
  int tid = blockIdx.x * 256 + threadIdx.x;   // B*HW threads
  int w = tid % WW;
  int h = (tid / WW) % HH;
  int b = tid / HWW;
  float sy = fminf(fmaxf(0.5f*h - 0.25f, 0.f), (float)(HD-1));
  float sx = fminf(fmaxf(0.5f*w - 0.25f, 0.f), (float)(HD-1));
  int ly = (int)sy; int hy = min(ly+1, HD-1); float wy = sy - (float)ly;
  int lx = (int)sx; int hx = min(lx+1, HD-1); float wx = sx - (float)lx;
  const float* xb = x + (size_t)b*CC*HWW + h*WW + w;
  const float* db = xd + (size_t)b*CC*HD*HD;
  float s = 0.f;
  for (int c = 0; c < CC; ++c) {
    const float* dc = db + c*HD*HD;
    float d00 = dc[ly*HD+lx], d01 = dc[ly*HD+hx];
    float d10 = dc[hy*HD+lx], d11 = dc[hy*HD+hx];
    float du = (1.f-wy)*((1.f-wx)*d00 + wx*d01) + wy*((1.f-wx)*d10 + wx*d11);
    s += fabsf(xb[(size_t)c*HWW] - du);
  }
  df[tid] = s;
  __shared__ float smn[256], smx[256];
  smn[threadIdx.x] = s; smx[threadIdx.x] = s;
  __syncthreads();
  for (int o = 128; o > 0; o >>= 1) {
    if (threadIdx.x < o) {
      smn[threadIdx.x] = fminf(smn[threadIdx.x], smn[threadIdx.x+o]);
      smx[threadIdx.x] = fmaxf(smx[threadIdx.x], smx[threadIdx.x+o]);
    }
    __syncthreads();
  }
  if (threadIdx.x == 0) {     // block spans one batch (HW % 256 == 0)
    atomicMin(&mm[b], __float_as_uint(smn[0]));
    atomicMax(&mm[BB+b], __float_as_uint(smx[0]));
  }
}

// ---------------- k3: GroupNorm stats ----------------
__global__ void k3_gn(const float* __restrict__ x, float* __restrict__ mean,
                      float* __restrict__ rstd) {
  int bg = blockIdx.x;                 // 0..127
  int b = bg >> 5, g = bg & 31;
  const float* p = x + (size_t)b*CC*HWW + (size_t)g*2*HWW;
  float s = 0.f, s2 = 0.f;
  for (int i = threadIdx.x; i < 2*HWW; i += 256) { float v = p[i]; s += v; s2 += v*v; }
  __shared__ float a0[256], a1[256];
  a0[threadIdx.x] = s; a1[threadIdx.x] = s2;
  __syncthreads();
  for (int o = 128; o > 0; o >>= 1) {
    if (threadIdx.x < o) { a0[threadIdx.x] += a0[threadIdx.x+o]; a1[threadIdx.x] += a1[threadIdx.x+o]; }
    __syncthreads();
  }
  if (threadIdx.x == 0) {
    const float inv = 1.f / (float)(2*HWW);
    float m = a0[0] * inv;
    float v = a1[0] * inv - m*m;
    mean[bg] = m;
    rstd[bg] = rsqrtf(v + 1e-5f);
  }
}

// ---------------- k4: similarity + adaptive top-k + GN apply ----------------
__global__ void k4_fuse(const float* __restrict__ x, const float* __restrict__ df,
                        const unsigned int* __restrict__ mm,
                        const float* __restrict__ mean, const float* __restrict__ rstd,
                        const float* __restrict__ gamma, const float* __restrict__ beta,
                        float* __restrict__ ef32, unsigned short* __restrict__ ebf) {
  int p = blockIdx.x * 256 + threadIdx.x;       // one pixel per thread
  int w = p % WW; int h = (p / WW) % HH; int b = p / HWW;
  const float* xb = x + (size_t)b*CC*HWW;
  int base = h*WW + w;

  int off[9]; bool ok[9];
  #pragma unroll
  for (int j = 0; j < 9; ++j) {
    int dy = j/3 - 1, dx = j%3 - 1;
    int hh2 = h + dy, ww2 = w + dx;
    ok[j]  = (hh2 >= 0 && hh2 < HH && ww2 >= 0 && ww2 < WW);
    off[j] = hh2*WW + ww2;
  }

  float dot[9], pn2[9], cn2 = 0.f;
  #pragma unroll
  for (int j = 0; j < 9; ++j) { dot[j] = 0.f; pn2[j] = 0.f; }
  for (int c = 0; c < CC; ++c) {
    const float* pc = xb + (size_t)c*HWW;
    float xc = pc[base];
    cn2 += xc*xc;
    #pragma unroll
    for (int j = 0; j < 9; ++j) {
      float pv = ok[j] ? pc[off[j]] : 0.f;
      dot[j] += xc*pv; pn2[j] += pv*pv;
    }
  }
  float cn = fmaxf(sqrtf(cn2), 1e-12f);
  float sim[9];
  #pragma unroll
  for (int j = 0; j < 9; ++j) sim[j] = dot[j] / (cn * fmaxf(sqrtf(pn2[j]), 1e-12f));

  float dmin = __uint_as_float(mm[b]);
  float dmax = __uint_as_float(mm[BB + b]);
  float rr  = (df[p] - dmin) / (dmax - dmin + 1e-8f);
  float dfp = rr * rr;
  float msk = (dfp > 0.3f) ? 1.f : 0.f;
  float connf = 1.f + fmaxf(rintf(dfp * 15.f), 0.f) * msk;
  bool process = connf > 1.f;
  int kk = (int)fminf(connf, 9.f);

  // stable descending argsort via pairwise rank with index tiebreak
  float wq[9]; float wsum = 0.f;
  #pragma unroll
  for (int i = 0; i < 9; ++i) {
    int rank = 0;
    #pragma unroll
    for (int j = 0; j < 9; ++j)
      rank += (sim[j] > sim[i]) || (sim[j] == sim[i] && j < i);
    float wv = (rank < kk) ? __expf(sim[i]) : 0.f;
    wq[i] = wv; wsum += wv;
  }
  float inv = 1.f / fmaxf(wsum, 1e-12f);
  #pragma unroll
  for (int i = 0; i < 9; ++i) wq[i] *= inv;

  const float* mb = mean + b*GG;
  const float* rb = rstd + b*GG;
  for (int c = 0; c < CC; ++c) {
    const float* pc = xb + (size_t)c*HWW;
    float xc = pc[base];
    float oc;
    if (process) {
      oc = 0.f;
      #pragma unroll
      for (int j = 0; j < 9; ++j) {
        float pv = ok[j] ? pc[off[j]] : 0.f;
        oc += wq[j] * pv;
      }
    } else oc = xc;
    int g = c >> 1;
    float nxv = (xc - mb[g]) * rb[g] * gamma[c] + beta[c];
    float e = oc + nxv;
    ef32[(size_t)p*64 + c] = e;
    ebf [(size_t)p*64 + c] = f2bf(e);
  }
}

// ---------------- k5: WMMA FFN (bf16 in, f32 acc) + residual ----------------
__global__ __launch_bounds__(256) void k5_ffn(const unsigned short* __restrict__ ebf,
                                              const float* __restrict__ ef32,
                                              const uint4* __restrict__ w1f,
                                              const uint4* __restrict__ w2f,
                                              const float* __restrict__ b1,
                                              const float* __restrict__ b2,
                                              float* __restrict__ out) {
  __shared__ __align__(16) char smem[8 * 4608];   // per-wave scratch
  int lane = threadIdx.x & 31;
  int wid  = threadIdx.x >> 5;
  int tile = blockIdx.x * 8 + wid;
  int n0   = tile * 16;                 // 16 consecutive pixels (HW % 16 == 0)
  int l15  = lane & 15;
  int half = lane >> 4;
  char* wb = smem + wid * 4608;

  // ---- GEMM1: h1[16,128] = E[16,64] x w1^T ----
  const uint4* rowq = (const uint4*)(ebf + (size_t)(n0 + l15) * 64);  // 8 uint4 per row
  v8f acc[8] = {};
  #pragma unroll
  for (int kt = 0; kt < 2; ++kt) {
    Frag a;
    a.q[0] = rowq[4*kt + half];         // vgprs 0..3: K = 32kt + 8*half + 0..7
    a.q[1] = rowq[4*kt + 2 + half];     // vgprs 4..7: K = 32kt + 16 + 8*half + 0..7
    #pragma unroll
    for (int ot = 0; ot < 8; ++ot) {
      int t = kt*8 + ot;
      Frag bf;
      bf.q[0] = w1f[(t*32 + lane)*2 + 0];
      bf.q[1] = w1f[(t*32 + lane)*2 + 1];
      acc[ot] = wmma_bf16(a.v, bf.v, acc[ot]);
    }
  }

  // bias + ReLU -> LDS as bf16 [m][128], row stride 272B (68 dwords, conflict-free)
  unsigned short* hls = (unsigned short*)wb;
  #pragma unroll
  for (int ot = 0; ot < 8; ++ot) {
    float bo = b1[ot*16 + l15];
    #pragma unroll
    for (int r = 0; r < 8; ++r) {
      int m = r + 8*half;
      float v = fmaxf(acc[ot][r] + bo, 0.f);
      hls[m*136 + ot*16 + l15] = f2bf(v);
    }
  }
  __syncthreads();

  // ---- GEMM2: y[16,64] = h1[16,128] x w2^T ----
  v8f acc2[4] = {};
  #pragma unroll
  for (int kt = 0; kt < 4; ++kt) {
    Frag a2;
    a2.q[0] = *(const uint4*)(wb + l15*272 + (16*kt +     4*half)*4);
    a2.q[1] = *(const uint4*)(wb + l15*272 + (16*kt + 8 + 4*half)*4);
    #pragma unroll
    for (int ct = 0; ct < 4; ++ct) {
      int t = kt*4 + ct;
      Frag bf;
      bf.q[0] = w2f[(t*32 + lane)*2 + 0];
      bf.q[1] = w2f[(t*32 + lane)*2 + 1];
      acc2[ct] = wmma_bf16(a2.v, bf.v, acc2[ct]);
    }
  }
  __syncthreads();

  // epilogue: + b2 + residual(enhanced f32), stage in LDS f32 [m][65] (padded)
  float* ols = (float*)wb;
  #pragma unroll
  for (int ct = 0; ct < 4; ++ct) {
    float bo = b2[ct*16 + l15];
    #pragma unroll
    for (int r = 0; r < 8; ++r) {
      int m = r + 8*half;
      int c = ct*16 + l15;
      ols[m*65 + c] = acc2[ct][r] + bo + ef32[(size_t)(n0 + m)*64 + c];
    }
  }
  __syncthreads();

  // coalesced planar stores: out[b, c, hw0 + m]
  int b   = n0 / HWW;
  int hw0 = n0 % HWW;
  float* ob = out + (size_t)b*CC*HWW + hw0;
  #pragma unroll
  for (int i = 0; i < 32; ++i) {
    int idx = i*32 + lane;
    int c = idx >> 4, m = idx & 15;
    ob[(size_t)c*HWW + m] = ols[m*65 + c];
  }
}

// ---------------- host launcher ----------------
extern "C" void kernel_launch(void* const* d_in, const int* in_sizes, int n_in,
                              void* d_out, int out_size, void* d_ws, size_t ws_size,
                              hipStream_t stream) {
  const float* x  = (const float*)d_in[0];
  const float* gw = (const float*)d_in[1];
  const float* gb = (const float*)d_in[2];
  const float* w1 = (const float*)d_in[3];
  const float* b1 = (const float*)d_in[4];
  const float* w2 = (const float*)d_in[5];
  const float* b2 = (const float*)d_in[6];
  char* ws = (char*)d_ws;

  // workspace layout (all offsets 256B-aligned)
  float*          xd   = (float*)(ws + 0);                 //  9,437,184 B
  float*          df   = (float*)(ws + 9437184);           //    589,824 B
  unsigned int*   mm   = (unsigned int*)(ws + 10027008);   //    min[4], max[4]
  float*          mean = (float*)(ws + 10027264);          //    512 B
  float*          rstd = (float*)(ws + 10027776);          //    512 B
  unsigned int*   w1f  = (unsigned int*)(ws + 10028288);   //  16,384 B
  unsigned int*   w2f  = (unsigned int*)(ws + 10044672);   //  16,384 B
  unsigned short* ebf  = (unsigned short*)(ws + 10061056); // 18,874,368 B
  float*          ef32 = (float*)(ws + 28935424);          // 37,748,736 B
  float*          out  = (float*)d_out;

  k0_init<<<1, 64, 0, stream>>>(mm);
  k0_pack<<<32, 256, 0, stream>>>(w1, w2, w1f, w2f);
  k1_down<<<(BB*CC*HD*HD + 255)/256, 256, 0, stream>>>(x, xd);
  k2_detail<<<NPIX/256, 256, 0, stream>>>(x, xd, df, mm);
  k3_gn<<<BB*GG, 256, 0, stream>>>(x, mean, rstd);
  k4_fuse<<<NPIX/256, 256, 0, stream>>>(x, df, mm, mean, rstd, gw, gb, ef32, ebf);
  k5_ffn<<<NPIX/16/8, 256, 0, stream>>>(ebf, ef32, (const uint4*)w1f, (const uint4*)w2f,
                                        b1, b2, out);
}
